// SE3Transformer_8272107012165
// MI455X (gfx1250) — matrix operand order repaired and checked by reading, not compiled, not run on hardware
//
#include <hip/hip_runtime.h>

#define NN 10000
#define EE 80000
#define CCH 32
#define HHD 8
#define LLAY 4
#define MIDD 32
#define OUTWW 1056

typedef _Float16 v16h __attribute__((ext_vector_type(16)));
typedef float    v8f  __attribute__((ext_vector_type(8)));

__device__ __forceinline__ float wsum32(float v) {
#pragma unroll
  for (int m = 16; m > 0; m >>= 1) v += __shfl_xor(v, m, 32);
  return v;
}

__device__ __forceinline__ unsigned fenc(float f) {
  unsigned u = __float_as_uint(f);
  return (u & 0x80000000u) ? ~u : (u | 0x80000000u);
}
__device__ __forceinline__ float fdec(unsigned e) {
  return __uint_as_float((e & 0x80000000u) ? (e ^ 0x80000000u) : ~e);
}

// ---- once: r = |rel_pos| ; xcur = node_feats ----
__global__ void k_prep(const float* __restrict__ rel, const float* __restrict__ nf,
                       float* __restrict__ r, float* __restrict__ xcur) {
  int t = blockIdx.x * blockDim.x + threadIdx.x;
  if (t < EE) {
    float a = rel[t * 3], b = rel[t * 3 + 1], c = rel[t * 3 + 2];
    r[t] = sqrtf(a * a + b * b + c * c);
  }
  if (t < NN * CCH) xcur[t] = nf[t];
}

// ---- once: build W3t in WMMA-B f16 layout for 5 weight sets ----
// B k order used here, contiguous and matching A (16-bit B 32x16): lane l -> N=l%16, K = j + (l/16)*16 for half j in [0,16)
__global__ void k_w3prep(const float* __restrict__ aWr3, const float* __restrict__ fWr3,
                         _Float16* __restrict__ B3) {
  int t = blockIdx.x * blockDim.x + threadIdx.x;
  const int TOT = 5 * 66 * 512;
  if (t >= TOT) return;
  int s = t / (66 * 512);
  int rem = t % (66 * 512);
  int blk = rem / 512;          // (ks*2 + nt)
  int w = rem % 512;
  int lane = w / 16, j = w % 16;
  int ks = blk >> 1, nt = blk & 1;
  int kk = ks * 32 + j + (lane / 16) * 16;   // p = m*33 + ii
  int m = kk / 33, ii = kk % 33;
  int o = nt * 16 + (lane % 16);
  const float* Ws = (s < 4) ? (aWr3 + s * (MIDD * OUTWW)) : fWr3;
  B3[t] = (_Float16)Ws[m * OUTWW + o * 33 + ii];
}

// ---- radial MLP stages 1-2: h2[e,32] (one wave per edge, lane = feature) ----
__global__ void k_radial(const float* __restrict__ r,
                         const float* __restrict__ W1, const float* __restrict__ b1,
                         const float* __restrict__ g1, const float* __restrict__ bb1,
                         const float* __restrict__ W2, const float* __restrict__ b2,
                         const float* __restrict__ g2, const float* __restrict__ bb2,
                         float* __restrict__ h2out) {
  __shared__ float sh[8][32];
  __shared__ float sW2[1024];
  int lane = threadIdx.x & 31;
  int wv = threadIdx.x >> 5;
  int e = blockIdx.x * 8 + wv;
  for (int t = threadIdx.x; t < 1024; t += 256) sW2[t] = W2[t];
  float rv = r[e];
  float h = rv * W1[lane] + b1[lane];
  float mu = wsum32(h) * (1.f / 32.f);
  float d = h - mu;
  float var = wsum32(d * d) * (1.f / 32.f);
  h = fmaxf(d * rsqrtf(var + 1e-5f) * g1[lane] + bb1[lane], 0.f);
  sh[wv][lane] = h;
  __syncthreads();
  float acc = b2[lane];
#pragma unroll
  for (int k = 0; k < 32; ++k) acc += sh[wv][k] * sW2[k * 32 + lane];
  mu = wsum32(acc) * (1.f / 32.f);
  d = acc - mu;
  var = wsum32(d * d) * (1.f / 32.f);
  float h2 = fmaxf(d * rsqrtf(var + 1e-5f) * g2[lane] + bb2[lane], 0.f);
  h2out[e * 32 + lane] = h2;
}

// ---- q = x @ Wq  (N x 16) ----
__global__ void k_q(const float* __restrict__ x, const float* __restrict__ Wq,
                    float* __restrict__ q) {
  int t = blockIdx.x * blockDim.x + threadIdx.x;
  if (t >= NN * 16) return;
  int n = t >> 4, o = t & 15;
  float s = 0.f;
#pragma unroll
  for (int c = 0; c < 32; ++c) s += x[n * 32 + c] * Wq[c * 16 + o];
  q[t] = s;
}

__global__ void k_initattn(unsigned* __restrict__ mmax, float* __restrict__ denom,
                           float* __restrict__ num) {
  int t = blockIdx.x * blockDim.x + threadIdx.x;
  if (t < NN * 8) { mmax[t] = 0u; denom[t] = 0.f; }
  if (t < NN * 16) num[t] = 0.f;
}

// ---- fused radial-W3 x f_src GEMM via WMMA: one wave per 16-edge tile ----
// A tile (16x32 f16): lane l -> M=l%16, halves j: K=j+(l/16)*16
// A[e, p] = h2[e, p/33] * fsrc[e, p%33] built on the fly (outer product, never in memory)
__global__ void __launch_bounds__(32)
k_kvgemm(const float* __restrict__ h2, const float* __restrict__ x,
         const float* __restrict__ r, const int* __restrict__ src,
         const _Float16* __restrict__ B3set,
         float* __restrict__ kvout,
         const int* __restrict__ dst, float* __restrict__ outAcc,
         int mode) {
  __shared__ float sh_h2[16][32];
  __shared__ float sh_f[16][34];
  int lane = threadIdx.x;
  int base = blockIdx.x * 16;
#pragma unroll
  for (int t = 0; t < 16; ++t) sh_h2[t][lane] = h2[(base + t) * 32 + lane];
#pragma unroll
  for (int t = 0; t < 16; ++t) {
    int se = src[base + t];
    sh_f[t][lane] = x[se * 32 + lane];
  }
  if (lane < 16) sh_f[lane][32] = r[base + lane];
  __syncthreads();
  int row = lane & 15, hi = lane >> 4;
  v8f acc0 = {}; v8f acc1 = {};
  for (int ks = 0; ks < 33; ++ks) {
    v16h a;
#pragma unroll
    for (int j = 0; j < 16; ++j) {
      int K = j + hi * 16;
      int kk = ks * 32 + K;
      int m = kk / 33;
      int ii = kk - m * 33;
      a[j] = (_Float16)(sh_h2[row][m] * sh_f[row][ii]);
    }
    v16h b0 = *(const v16h*)(B3set + (ks * 2 + 0) * 512 + lane * 16);
    v16h b1 = *(const v16h*)(B3set + (ks * 2 + 1) * 512 + lane * 16);
    acc0 = __builtin_amdgcn_wmma_f32_16x16x32_f16(false, a, false, b0, (short)0, acc0, false, false);
    acc1 = __builtin_amdgcn_wmma_f32_16x16x32_f16(false, a, false, b1, (short)0, acc1, false, false);
  }
  // D layout: lane -> N = lane%16 ; VGPR v -> M = v + 8*(lane/16)
  if (mode == 0) {
#pragma unroll
    for (int v = 0; v < 8; ++v) {
      int e = base + v + hi * 8;
      kvout[e * 32 + row] = acc0[v];
      kvout[e * 32 + 16 + row] = acc1[v];
    }
  } else {
#pragma unroll
    for (int v = 0; v < 8; ++v) {
      int e = base + v + hi * 8;
      int dn = dst[e];
      atomicAdd(outAcc + dn * 32 + row, acc0[v]);
      atomicAdd(outAcc + dn * 32 + 16 + row, acc1[v]);
    }
  }
}

// ---- logits + segment max (encoded uint atomicMax) ----
__global__ void k_logits(const float* __restrict__ kv, const float* __restrict__ q,
                         const int* __restrict__ dst, float* __restrict__ logits,
                         unsigned* __restrict__ mmax) {
  int t = blockIdx.x * blockDim.x + threadIdx.x;
  if (t >= EE * 8) return;
  int e = t >> 3, h = t & 7;
  int dn = dst[e];
  float l = (kv[e * 32 + 2 * h] * q[dn * 16 + 2 * h] +
             kv[e * 32 + 2 * h + 1] * q[dn * 16 + 2 * h + 1]) * 0.70710678118654752f;
  logits[t] = l;
  atomicMax(mmax + dn * 8 + h, fenc(l));
}

// ---- z = exp(logit - max); accumulate denom and z*v numerators ----
__global__ void k_passb(const float* __restrict__ kv, const float* __restrict__ logits,
                        const int* __restrict__ dst, const unsigned* __restrict__ mmax,
                        float* __restrict__ denom, float* __restrict__ num) {
  int t = blockIdx.x * blockDim.x + threadIdx.x;
  if (t >= EE * 8) return;
  int e = t >> 3, h = t & 7;
  int dn = dst[e];
  float z = expf(logits[t] - fdec(mmax[dn * 8 + h]));
  atomicAdd(denom + dn * 8 + h, z);
  atomicAdd(num + dn * 16 + 2 * h,     z * kv[e * 32 + 16 + 2 * h]);
  atomicAdd(num + dn * 16 + 2 * h + 1, z * kv[e * 32 + 16 + 2 * h + 1]);
}

// ---- attn/denom -> concat proj -> NormSE3, in-place x update (wave per node) ----
__global__ void k_node(const float* __restrict__ num, const float* __restrict__ denom,
                       const float* __restrict__ Wp, const float* __restrict__ g,
                       const float* __restrict__ b, float* __restrict__ x) {
  __shared__ float sa[8][16];
  __shared__ float sx[8][32];
  __shared__ float sWp[1536];
  int lane = threadIdx.x & 31, wv = threadIdx.x >> 5;
  int n = blockIdx.x * 8 + wv;
  for (int t = threadIdx.x; t < 1536; t += 256) sWp[t] = Wp[t];
  sx[wv][lane] = x[n * 32 + lane];
  if (lane < 16) {
    float dv = denom[n * 8 + (lane >> 1)];
    sa[wv][lane] = (dv > 0.f) ? num[n * 16 + lane] / dv : 0.f;
  }
  __syncthreads();
  float y = 0.f;
#pragma unroll
  for (int j = 0; j < 16; ++j) y += sa[wv][j] * sWp[j * 32 + lane];
#pragma unroll
  for (int j = 0; j < 32; ++j) y += sx[wv][j] * sWp[(16 + j) * 32 + lane];
  float nv = sqrtf(y * y + 1e-12f);
  float mu = wsum32(nv) * (1.f / 32.f);
  float d = nv - mu;
  float var = wsum32(d * d) * (1.f / 32.f);
  float ln = fmaxf(d * rsqrtf(var + 1e-5f) * g[lane] + b[lane], 0.f);
  x[n * 32 + lane] = y * ln / nv;
}

// ---- out = x @ fWself (initializes d_out before edge atomics) ----
__global__ void k_selfout(const float* __restrict__ x, const float* __restrict__ Ws,
                          float* __restrict__ out) {
  int t = blockIdx.x * blockDim.x + threadIdx.x;
  if (t >= NN * 32) return;
  int n = t >> 5, c = t & 31;
  float s = 0.f;
#pragma unroll
  for (int j = 0; j < 32; ++j) s += x[n * 32 + j] * Ws[j * 32 + c];
  out[t] = s;
}

extern "C" void kernel_launch(void* const* d_in, const int* in_sizes, int n_in,
                              void* d_out, int out_size, void* d_ws, size_t ws_size,
                              hipStream_t stream) {
  const float* node_feats = (const float*)d_in[0];
  const float* rel_pos    = (const float*)d_in[1];
  const int*   src        = (const int*)d_in[2];
  const int*   dst        = (const int*)d_in[3];
  const float* aWr1 = (const float*)d_in[4];
  const float* abr1 = (const float*)d_in[5];
  const float* ag1  = (const float*)d_in[6];
  const float* ab1  = (const float*)d_in[7];
  const float* aWr2 = (const float*)d_in[8];
  const float* abr2 = (const float*)d_in[9];
  const float* ag2  = (const float*)d_in[10];
  const float* ab2  = (const float*)d_in[11];
  const float* aWr3 = (const float*)d_in[12];
  const float* aWq  = (const float*)d_in[13];
  const float* aWpr = (const float*)d_in[14];
  const float* ng   = (const float*)d_in[15];
  const float* nb   = (const float*)d_in[16];
  const float* fWr1 = (const float*)d_in[17];
  const float* fbr1 = (const float*)d_in[18];
  const float* fg1  = (const float*)d_in[19];
  const float* fb1  = (const float*)d_in[20];
  const float* fWr2 = (const float*)d_in[21];
  const float* fbr2 = (const float*)d_in[22];
  const float* fg2  = (const float*)d_in[23];
  const float* fb2  = (const float*)d_in[24];
  const float* fWr3 = (const float*)d_in[25];
  const float* fWself = (const float*)d_in[26];
  float* out = (float*)d_out;

  float* ws = (float*)d_ws;
  size_t off = 0;
  auto carve = [&](size_t nf) {
    float* p = ws + off;
    off += (nf + 15) & ~(size_t)15;   // 64B-aligned segments
    return p;
  };
  float*    r      = carve(EE);
  float*    xcur   = carve((size_t)NN * 32);
  float*    h2     = carve((size_t)EE * 32);
  float*    kv     = carve((size_t)EE * 32);
  float*    logits = carve((size_t)EE * 8);
  float*    q      = carve((size_t)NN * 16);
  unsigned* mmax   = (unsigned*)carve((size_t)NN * 8);
  float*    denom  = carve((size_t)NN * 8);
  float*    num    = carve((size_t)NN * 16);
  _Float16* B3     = (_Float16*)carve(5 * 66 * 512 / 2);

  dim3 b256(256);
  k_prep<<<(NN * 32 + 255) / 256, b256, 0, stream>>>(rel_pos, node_feats, r, xcur);
  k_w3prep<<<(5 * 66 * 512 + 255) / 256, b256, 0, stream>>>(aWr3, fWr3, B3);

  for (int i = 0; i < LLAY; ++i) {
    k_radial<<<EE / 8, b256, 0, stream>>>(r, aWr1 + i * 32, abr1 + i * 32, ag1 + i * 32,
                                          ab1 + i * 32, aWr2 + i * 1024, abr2 + i * 32,
                                          ag2 + i * 32, ab2 + i * 32, h2);
    k_q<<<(NN * 16 + 255) / 256, b256, 0, stream>>>(xcur, aWq + i * 512, q);
    k_initattn<<<(NN * 16 + 255) / 256, b256, 0, stream>>>(mmax, denom, num);
    k_kvgemm<<<EE / 16, 32, 0, stream>>>(h2, xcur, r, src, B3 + (size_t)i * 66 * 512,
                                         kv, dst, nullptr, 0);
    k_logits<<<(EE * 8 + 255) / 256, b256, 0, stream>>>(kv, q, dst, logits, mmax);
    k_passb<<<(EE * 8 + 255) / 256, b256, 0, stream>>>(kv, logits, dst, mmax, denom, num);
    k_node<<<NN / 8, b256, 0, stream>>>(num, denom, aWpr + i * 1536, ng + i * 32,
                                        nb + i * 32, xcur);
  }

  // final ConvSE3 (pool=True, self_interaction=True)
  k_radial<<<EE / 8, b256, 0, stream>>>(r, fWr1, fbr1, fg1, fb1, fWr2, fbr2, fg2, fb2, h2);
  k_selfout<<<(NN * 32 + 255) / 256, b256, 0, stream>>>(xcur, fWself, out);
  k_kvgemm<<<EE / 16, 32, 0, stream>>>(h2, xcur, r, src, B3 + (size_t)4 * 66 * 512,
                                       nullptr, dst, out, 1);
}
